// InterCLoss_24292335026250
// MI455X (gfx1250) — matrix operand order repaired and checked by reading, not compiled, not run on hardware
//
#include <hip/hip_runtime.h>
#include <hip/hip_bf16.h>
#include <math.h>

typedef __attribute__((ext_vector_type(16))) __bf16 v16bf;
typedef __attribute__((ext_vector_type(8)))  float  v8f;

#define BS   4096
#define DDIM 512

// ---------- helpers ----------
__device__ __forceinline__ unsigned short f2bf(float f) {
    unsigned int u = __float_as_uint(f);
    u += 0x7fffu + ((u >> 16) & 1u);          // round-to-nearest-even
    return (unsigned short)(u >> 16);
}
__device__ __forceinline__ float bf2f(unsigned short h) {
    return __uint_as_float(((unsigned int)h) << 16);
}
// monotone float <-> orderable uint (for atomicMax on floats incl. negatives)
__device__ __forceinline__ unsigned int f2ord(float f) {
    unsigned int u = __float_as_uint(f);
    return (u & 0x80000000u) ? ~u : (u | 0x80000000u);
}
__device__ __forceinline__ float ord2f(unsigned int u) {
    unsigned int b = (u & 0x80000000u) ? (u ^ 0x80000000u) : ~u;
    return __uint_as_float(b);
}

union V16 { v16bf bf; uint4 u4[2]; };

// ---------- kernel 0: zero the statistics ----------
__global__ void init_stats_kernel(unsigned int* rowMaxU, unsigned int* colMaxU,
                                  float* rowZ, float* rowT, float* colZ, float* colT,
                                  float* Ssum) {
    int i = blockIdx.x * blockDim.x + threadIdx.x;
    if (i < BS) {
        rowMaxU[i] = 0u;   // encoded -inf
        colMaxU[i] = 0u;
        rowZ[i] = 0.f; rowT[i] = 0.f; colZ[i] = 0.f; colT[i] = 0.f;
    }
    if (i == 0) *Ssum = 0.f;
}

// ---------- kernel 1: L2-normalize rows + split into bf16 hi/lo ----------
__global__ void norm_split_kernel(const float* __restrict__ img, const float* __restrict__ tab,
                                  unsigned short* __restrict__ imgHi, unsigned short* __restrict__ imgLo,
                                  unsigned short* __restrict__ tabHi, unsigned short* __restrict__ tabLo) {
    const int row = blockIdx.x;
    const float* src = blockIdx.y ? tab : img;
    unsigned short* hi = blockIdx.y ? tabHi : imgHi;
    unsigned short* lo = blockIdx.y ? tabLo : imgLo;

    const int t = threadIdx.x;                        // 128 threads, 4 floats each
    float4 x = ((const float4*)(src + (size_t)row * DDIM))[t];
    float ss = x.x * x.x + x.y * x.y + x.z * x.z + x.w * x.w;
    #pragma unroll
    for (int m = 1; m < 32; m <<= 1) ss += __shfl_xor(ss, m, 32);
    __shared__ float wsum[4];
    if ((t & 31) == 0) wsum[t >> 5] = ss;
    __syncthreads();
    float tot = wsum[0] + wsum[1] + wsum[2] + wsum[3];
    float inv = 1.0f / fmaxf(sqrtf(tot), 1e-12f);

    float xn[4] = { x.x * inv, x.y * inv, x.z * inv, x.w * inv };
    #pragma unroll
    for (int e = 0; e < 4; ++e) {
        unsigned short h = f2bf(xn[e]);
        float r = xn[e] - bf2f(h);
        size_t idx = (size_t)row * DDIM + t * 4 + e;
        hi[idx] = h;
        lo[idx] = f2bf(r);
    }
}

// ---------- fragment loaders (layouts per CDNA5 ISA 7.12.2) ----------
__device__ __forceinline__ void load_a(V16& hi, V16& lo,
                                       const unsigned short* aH, const unsigned short* aL,
                                       int k0, int half) {
    // A (16x32, MxK): lanes 0-15 hold K = k0+[0..7], k0+[16..23]; lanes 16-31 shift by 8
    const int a0 = k0 + half * 8;
    const int a1 = k0 + 16 + half * 8;
    hi.u4[0] = *(const uint4*)(aH + a0);  hi.u4[1] = *(const uint4*)(aH + a1);
    lo.u4[0] = *(const uint4*)(aL + a0);  lo.u4[1] = *(const uint4*)(aL + a1);
}
__device__ __forceinline__ void load_b(V16& hi, V16& lo,
                                       const unsigned short* bH, const unsigned short* bL,
                                       int k0, int half) {
    // B (32x16, KxN): lane holds column N=l16, K = k0+[0..7], k0+[16..23]; lanes 16-31 shift by 8
    const int b0 = k0 + half * 8;
    const int b1 = k0 + 16 + half * 8;
    hi.u4[0] = *(const uint4*)(bH + b0);  hi.u4[1] = *(const uint4*)(bH + b1);
    lo.u4[0] = *(const uint4*)(bL + b0);  lo.u4[1] = *(const uint4*)(bL + b1);
}

// ---------- split-bf16 WMMA, 2x2 register-blocked 32x32 tile per wave ----------
__device__ __forceinline__ void gemm_2x2(v8f c[2][2],
                                         const unsigned short* __restrict__ imgHi,
                                         const unsigned short* __restrict__ imgLo,
                                         const unsigned short* __restrict__ tabHi,
                                         const unsigned short* __restrict__ tabLo,
                                         int iBase, int jBase, int lane) {
    const int l16  = lane & 15;
    const int half = (lane >> 4) & 1;
    const unsigned short* aH0 = imgHi + (size_t)(iBase + l16) * DDIM;
    const unsigned short* aL0 = imgLo + (size_t)(iBase + l16) * DDIM;
    const unsigned short* bH0 = tabHi + (size_t)(jBase + l16) * DDIM;
    const unsigned short* bL0 = tabLo + (size_t)(jBase + l16) * DDIM;
    const unsigned short* aH1 = aH0 + (size_t)16 * DDIM;
    const unsigned short* aL1 = aL0 + (size_t)16 * DDIM;
    const unsigned short* bH1 = bH0 + (size_t)16 * DDIM;
    const unsigned short* bL1 = bL0 + (size_t)16 * DDIM;

    for (int k0 = 0; k0 < DDIM; k0 += 32) {
        V16 ah[2], al[2], bh[2], bl[2];
        load_a(ah[0], al[0], aH0, aL0, k0, half);
        load_a(ah[1], al[1], aH1, aL1, k0, half);
        load_b(bh[0], bl[0], bH0, bL0, k0, half);
        load_b(bh[1], bl[1], bH1, bL1, k0, half);
        #pragma unroll
        for (int ti = 0; ti < 2; ++ti) {
            #pragma unroll
            for (int tj = 0; tj < 2; ++tj) {
                c[ti][tj] = __builtin_amdgcn_wmma_f32_16x16x32_bf16(false, ah[ti].bf, false, bh[tj].bf, (short)0, c[ti][tj], false, false);
                c[ti][tj] = __builtin_amdgcn_wmma_f32_16x16x32_bf16(false, ah[ti].bf, false, bl[tj].bf, (short)0, c[ti][tj], false, false);
                c[ti][tj] = __builtin_amdgcn_wmma_f32_16x16x32_bf16(false, al[ti].bf, false, bh[tj].bf, (short)0, c[ti][tj], false, false);
            }
        }
    }
}

// ---------- kernel 2 (pass 1): row / column maxima of A = s * imgN @ tabN^T ----------
__global__ void pass1_max_kernel(const unsigned short* __restrict__ imgHi, const unsigned short* __restrict__ imgLo,
                                 const unsigned short* __restrict__ tabHi, const unsigned short* __restrict__ tabLo,
                                 const float* __restrict__ scalePtr,
                                 unsigned int* __restrict__ rowMaxU, unsigned int* __restrict__ colMaxU) {
    const int wave = threadIdx.x >> 5;
    const int lane = threadIdx.x & 31;
    const int iBase = (blockIdx.y * 8 + wave) * 32;
    const int jBase = blockIdx.x * 32;
    const float s = expf(scalePtr[0]);

    v8f c[2][2] = {};
    gemm_2x2(c, imgHi, imgLo, tabHi, tabLo, iBase, jBase, lane);

    const int l16  = lane & 15;
    const int half = (lane >> 4) & 1;

    // row maxima: combine both j-tiles, reduce over the 16-lane row group
    #pragma unroll
    for (int ti = 0; ti < 2; ++ti) {
        #pragma unroll
        for (int v = 0; v < 8; ++v) {
            float rm = fmaxf(s * c[ti][0][v], s * c[ti][1][v]);
            rm = fmaxf(rm, __shfl_xor(rm, 1, 32));
            rm = fmaxf(rm, __shfl_xor(rm, 2, 32));
            rm = fmaxf(rm, __shfl_xor(rm, 4, 32));
            rm = fmaxf(rm, __shfl_xor(rm, 8, 32));
            if (l16 == 0) atomicMax(&rowMaxU[iBase + ti * 16 + v + half * 8], f2ord(rm));
        }
    }
    // column maxima: combine both i-tiles and both M halves
    #pragma unroll
    for (int tj = 0; tj < 2; ++tj) {
        float colm = -INFINITY;
        #pragma unroll
        for (int ti = 0; ti < 2; ++ti)
            #pragma unroll
            for (int v = 0; v < 8; ++v)
                colm = fmaxf(colm, s * c[ti][tj][v]);
        colm = fmaxf(colm, __shfl_xor(colm, 16, 32));
        if (half == 0) atomicMax(&colMaxU[jBase + tj * 16 + l16], f2ord(colm));
    }
}

// ---------- kernel 3 (pass 2): softmax sums Z, T (rows & cols) and total S ----------
__global__ void pass2_sum_kernel(const unsigned short* __restrict__ imgHi, const unsigned short* __restrict__ imgLo,
                                 const unsigned short* __restrict__ tabHi, const unsigned short* __restrict__ tabLo,
                                 const float* __restrict__ scalePtr,
                                 const unsigned int* __restrict__ rowMaxU, const unsigned int* __restrict__ colMaxU,
                                 float* __restrict__ rowZ, float* __restrict__ rowT,
                                 float* __restrict__ colZ, float* __restrict__ colT,
                                 float* __restrict__ Ssum) {
    const int wave = threadIdx.x >> 5;
    const int lane = threadIdx.x & 31;
    const int iBase = (blockIdx.y * 8 + wave) * 32;
    const int jBase = blockIdx.x * 32;
    const float s = expf(scalePtr[0]);

    v8f c[2][2] = {};
    gemm_2x2(c, imgHi, imgLo, tabHi, tabLo, iBase, jBase, lane);

    const int l16  = lane & 15;
    const int half = (lane >> 4) & 1;
    float mc[2] = { ord2f(colMaxU[jBase + l16]), ord2f(colMaxU[jBase + 16 + l16]) };
    float cz[2] = { 0.f, 0.f }, ct[2] = { 0.f, 0.f };
    float stot = 0.f;

    #pragma unroll
    for (int ti = 0; ti < 2; ++ti) {
        #pragma unroll
        for (int v = 0; v < 8; ++v) {
            float mr = ord2f(rowMaxU[iBase + ti * 16 + v + half * 8]);
            float rz = 0.f, rt = 0.f;
            #pragma unroll
            for (int tj = 0; tj < 2; ++tj) {
                float a  = s * c[ti][tj][v];      // A[iBase+16ti+v+8*half][jBase+16tj+l16]
                float er = expf(a - mr);
                rz += er; rt += er * a;
                float ec = expf(a - mc[tj]);
                cz[tj] += ec; ct[tj] += ec * a;
                stot += a;
            }
            #pragma unroll
            for (int m = 1; m < 16; m <<= 1) { rz += __shfl_xor(rz, m, 32); rt += __shfl_xor(rt, m, 32); }
            if (l16 == 0) {
                atomicAdd(&rowZ[iBase + ti * 16 + v + half * 8], rz);
                atomicAdd(&rowT[iBase + ti * 16 + v + half * 8], rt);
            }
        }
    }
    #pragma unroll
    for (int tj = 0; tj < 2; ++tj) {
        cz[tj] += __shfl_xor(cz[tj], 16, 32);
        ct[tj] += __shfl_xor(ct[tj], 16, 32);
        if (half == 0) {
            atomicAdd(&colZ[jBase + tj * 16 + l16], cz[tj]);
            atomicAdd(&colT[jBase + tj * 16 + l16], ct[tj]);
        }
    }
    #pragma unroll
    for (int m = 1; m < 32; m <<= 1) stot += __shfl_xor(stot, m, 32);
    if (lane == 0) atomicAdd(Ssum, stot);
}

// ---------- kernel 4: assemble the scalar loss ----------
__global__ void finalize_kernel(const unsigned int* __restrict__ rowMaxU, const unsigned int* __restrict__ colMaxU,
                                const float* __restrict__ rowZ, const float* __restrict__ rowT,
                                const float* __restrict__ colZ, const float* __restrict__ colT,
                                const float* __restrict__ Ssum, float* __restrict__ out) {
    float rA = 0.f, rE = 0.f, cA = 0.f, cE = 0.f;
    for (int i = threadIdx.x; i < BS; i += 256) {
        float m  = ord2f(rowMaxU[i]); float z = rowZ[i]; float lz = logf(z);
        rA += m + lz;
        rE += rowT[i] / z - m - lz;
        m  = ord2f(colMaxU[i]); z = colZ[i]; lz = logf(z);
        cA += m + lz;
        cE += colT[i] / z - m - lz;
    }
    __shared__ float sh[4][256];
    sh[0][threadIdx.x] = rA; sh[1][threadIdx.x] = rE;
    sh[2][threadIdx.x] = cA; sh[3][threadIdx.x] = cE;
    __syncthreads();
    for (int off = 128; off > 0; off >>= 1) {
        if (threadIdx.x < off)
            #pragma unroll
            for (int q = 0; q < 4; ++q) sh[q][threadIdx.x] += sh[q][threadIdx.x + off];
        __syncthreads();
    }
    if (threadIdx.x == 0) {
        const float bs = (float)BS, lnbs = logf((float)BS);
        const float S = *Ssum;
        float l1 = (-bs * bs * lnbs - S + bs * sh[0][0]) / (bs * bs * bs);
        float l2 = (sh[1][0] + bs * lnbs) / (bs * bs);
        float l3 = (-bs * bs * lnbs - S + bs * sh[2][0]) / (bs * bs * bs);
        float l4 = (sh[3][0] + bs * lnbs) / (bs * bs);
        out[0] = 0.25f * (l1 + l2 + l3 + l4);
    }
}

extern "C" void kernel_launch(void* const* d_in, const int* in_sizes, int n_in,
                              void* d_out, int out_size, void* d_ws, size_t ws_size,
                              hipStream_t stream) {
    const float* img   = (const float*)d_in[0];
    const float* tab   = (const float*)d_in[1];
    // d_in[2] = label: provably irrelevant (softmax of a constant row is uniform for any labels)
    const float* scale = (const float*)d_in[3];
    float* out = (float*)d_out;

    char* ws = (char*)d_ws;
    size_t o = 0;
    const size_t matB = (size_t)BS * DDIM * sizeof(unsigned short); // 4 MiB
    unsigned short* imgHi = (unsigned short*)(ws + o); o += matB;
    unsigned short* imgLo = (unsigned short*)(ws + o); o += matB;
    unsigned short* tabHi = (unsigned short*)(ws + o); o += matB;
    unsigned short* tabLo = (unsigned short*)(ws + o); o += matB;
    unsigned int* rowMaxU = (unsigned int*)(ws + o);   o += BS * 4;
    unsigned int* colMaxU = (unsigned int*)(ws + o);   o += BS * 4;
    float* rowZ = (float*)(ws + o); o += BS * 4;
    float* rowT = (float*)(ws + o); o += BS * 4;
    float* colZ = (float*)(ws + o); o += BS * 4;
    float* colT = (float*)(ws + o); o += BS * 4;
    float* Ssum = (float*)(ws + o); o += 256;

    init_stats_kernel<<<(BS + 255) / 256, 256, 0, stream>>>(rowMaxU, colMaxU, rowZ, rowT, colZ, colT, Ssum);
    norm_split_kernel<<<dim3(BS, 2), 128, 0, stream>>>(img, tab, imgHi, imgLo, tabHi, tabLo);

    dim3 grid(BS / 32, BS / 32 / 8);   // (128, 16): 8 waves/block, one 32x32 region per wave
    pass1_max_kernel<<<grid, 256, 0, stream>>>(imgHi, imgLo, tabHi, tabLo, scale, rowMaxU, colMaxU);
    pass2_sum_kernel<<<grid, 256, 0, stream>>>(imgHi, imgLo, tabHi, tabLo, scale,
                                               rowMaxU, colMaxU, rowZ, rowT, colZ, colT, Ssum);
    finalize_kernel<<<1, 256, 0, stream>>>(rowMaxU, colMaxU, rowZ, rowT, colZ, colT, Ssum, out);
}
